// HVFeatureNet_41326175322615
// MI455X (gfx1250) — matrix operand order repaired and checked by reading, not compiled, not run on hardware
//
#include <hip/hip_runtime.h>
#include <hip/hip_fp16.h>

typedef __attribute__((ext_vector_type(16))) _Float16 v16h;
typedef __attribute__((ext_vector_type(8)))  _Float16 v8h;
typedef __attribute__((ext_vector_type(8)))  float    v8f;

#define NPTS 300000
#define KDIM 384           // 3 * 128 input features to AVFEO
#define V0S 60000
#define V1S 30000
#define V2S 15000
#define V3S 120000
#define HH 512
#define WW 512

// ---------------- workspace layout (bytes) ----------------
#define OFF_SUMS0 ((size_t)0)
#define OFF_SUMS1 (OFF_SUMS0 + (size_t)V0S * 16)
#define OFF_SUMS2 (OFF_SUMS1 + (size_t)V1S * 16)
#define OFF_SUMS3 (OFF_SUMS2 + (size_t)V2S * 16)
#define OFF_CNT0  (OFF_SUMS3 + (size_t)V3S * 16)
#define OFF_CNT1  (OFF_CNT0 + (size_t)V0S * 4)
#define OFF_CNT2  (OFF_CNT1 + (size_t)V1S * 4)
#define OFF_CNT3  (OFF_CNT2 + (size_t)V2S * 4)
#define OFF_VMAX0 (OFF_CNT3 + (size_t)V3S * 4)
#define OFF_VMAX1 (OFF_VMAX0 + (size_t)V0S * 64 * 4)
#define OFF_VMAX2 (OFF_VMAX1 + (size_t)V1S * 64 * 4)
#define OFF_VMAX3 (OFF_VMAX2 + (size_t)V2S * 64 * 4)
#define OFF_PACKB (OFF_VMAX3 + (size_t)V3S * 128 * 4)      // = 92,820,000 (32B aligned)
#define OFF_FINAL (OFF_PACKB + (size_t)8 * 12 * 32 * 16 * 2) // 98,304 B of packed B frags
#define ZERO_BYTES OFF_PACKB

// ---------------- utility: zero fill ----------------
__global__ void hv_zero_u4(uint4* __restrict__ p, long n16) {
    long i = (long)blockIdx.x * blockDim.x + threadIdx.x;
    if (i < n16) { uint4 z; z.x = z.y = z.z = z.w = 0u; p[i] = z; }
}

// ---------------- pack W_pt2 (fp32 [384,128]) into WMMA-B fragments (f16) ----
// frag (colblock c in 0..7, kstep t in 0..11): lane holds 16 contiguous halves.
// B layout (16-bit, 32x16): lanes 0-15 = column (lane&15), K = t*32 + {0..7,16..23};
// lanes 16-31 = same columns, K = t*32 + {8..15,24..31}.
__global__ void hv_pack_b(const float* __restrict__ W, _Float16* __restrict__ packB) {
    int t = blockIdx.x * blockDim.x + threadIdx.x;       // 8*12*32*16 = 49152
    if (t >= 8 * 12 * 32 * 16) return;
    int j    = t & 15;
    int lane = (t >> 4) & 31;
    int ft   = t >> 9;           // fragment index = c*12 + tk
    int tk   = ft % 12;
    int c    = ft / 12;
    int k = tk * 32 + ((lane >> 4) ? 8 : 0) + (j & 7) + ((j >> 3) << 4);
    int n = c * 16 + (lane & 15);
    packB[t] = (_Float16)W[k * 128 + n];
}

// ---------------- segment sums + counts for all 4 scales ----------------
__global__ void hv_seg_sums(const float4* __restrict__ pts, const int* __restrict__ pv,
                            float* __restrict__ s0, float* __restrict__ c0,
                            float* __restrict__ s1, float* __restrict__ c1,
                            float* __restrict__ s2, float* __restrict__ c2,
                            float* __restrict__ s3, float* __restrict__ c3) {
    int p = blockIdx.x * blockDim.x + threadIdx.x;
    if (p >= NPTS) return;
    float4 pt = pts[p];
    int v;
    v = pv[p];
    atomicAdd(&s0[v * 4 + 0], pt.x); atomicAdd(&s0[v * 4 + 1], pt.y);
    atomicAdd(&s0[v * 4 + 2], pt.z); atomicAdd(&s0[v * 4 + 3], pt.w);
    atomicAdd(&c0[v], 1.0f);
    v = pv[NPTS + p];
    atomicAdd(&s1[v * 4 + 0], pt.x); atomicAdd(&s1[v * 4 + 1], pt.y);
    atomicAdd(&s1[v * 4 + 2], pt.z); atomicAdd(&s1[v * 4 + 3], pt.w);
    atomicAdd(&c1[v], 1.0f);
    v = pv[2 * NPTS + p];
    atomicAdd(&s2[v * 4 + 0], pt.x); atomicAdd(&s2[v * 4 + 1], pt.y);
    atomicAdd(&s2[v * 4 + 2], pt.z); atomicAdd(&s2[v * 4 + 3], pt.w);
    atomicAdd(&c2[v], 1.0f);
    v = pv[3 * NPTS + p];
    atomicAdd(&s3[v * 4 + 0], pt.x); atomicAdd(&s3[v * 4 + 1], pt.y);
    atomicAdd(&s3[v * 4 + 2], pt.z); atomicAdd(&s3[v * 4 + 3], pt.w);
    atomicAdd(&c3[v], 1.0f);
}

// ---------------- fused AVFE: pf1, att, af, sf -> finalh + atomic max ----------
__device__ __forceinline__ void avfe_scale(int p, int j,
        float px, float py, float pz, float pw, float pf1,
        int vidx, const float* __restrict__ sums, const float* __restrict__ cnts,
        unsigned* __restrict__ vmax, const float* __restrict__ Watt1L,
        _Float16* __restrict__ frow) {
    float c = fmaxf(cnts[vidx], 1.0f);
    float inv = 1.0f / c;
    float mx = sums[vidx * 4 + 0] * inv, my = sums[vidx * 4 + 1] * inv;
    float mz = sums[vidx * 4 + 2] * inv, mw = sums[vidx * 4 + 3] * inv;
    float att[8] = {px - mx, py - my, pz - mz, pw, mx, my, mz, mw};
    float af = 0.0f;
#pragma unroll
    for (int q = 0; q < 8; q++) af += att[q] * Watt1L[q * 64 + j];
    af = fmaxf(af, 0.0f);
    float sf = pf1 * af;                    // >= 0: bit pattern order == uint order
    frow[j] = (_Float16)sf;
    atomicMax(&vmax[(long)vidx * 64 + j], __float_as_uint(sf));
}

__global__ void __launch_bounds__(256) hv_avfe(
        const float* __restrict__ pts, const int* __restrict__ pv,
        const float* __restrict__ Wpt1, const float* __restrict__ Watt1,
        const float* __restrict__ s0, const float* __restrict__ c0,
        const float* __restrict__ s1, const float* __restrict__ c1,
        const float* __restrict__ s2, const float* __restrict__ c2,
        unsigned* __restrict__ vm0, unsigned* __restrict__ vm1, unsigned* __restrict__ vm2,
        _Float16* __restrict__ finalh) {
    __shared__ float sW[4 * 64 + 8 * 64];   // W_pt1 then W_att1
    for (int i = threadIdx.x; i < 768; i += 256)
        sW[i] = (i < 256) ? Wpt1[i] : Watt1[i - 256];
    __syncthreads();
    long tid = (long)blockIdx.x * 256 + threadIdx.x;    // exactly NPTS*64 threads
    int p = (int)(tid >> 6);
    int j = (int)(tid & 63);
    float px = pts[4 * p + 0], py = pts[4 * p + 1];
    float pz = pts[4 * p + 2], pw = pts[4 * p + 3];
    float pf1 = px * sW[0 * 64 + j] + py * sW[1 * 64 + j] +
                pz * sW[2 * 64 + j] + pw * sW[3 * 64 + j];
    pf1 = fmaxf(pf1, 0.0f);
    const float* Watt1L = sW + 256;
    _Float16* frow = finalh + (long)p * KDIM;
    avfe_scale(p, j, px, py, pz, pw, pf1, pv[p],            s0, c0, vm0, Watt1L, frow + 0);
    avfe_scale(p, j, px, py, pz, pw, pf1, pv[NPTS + p],     s1, c1, vm1, Watt1L, frow + 128);
    avfe_scale(p, j, px, py, pz, pw, pf1, pv[2 * NPTS + p], s2, c2, vm2, Watt1L, frow + 256);
}

// ---------------- gather per-voxel maxima back per point --------------------
__global__ void __launch_bounds__(256) hv_gather(
        const int* __restrict__ pv,
        const unsigned* __restrict__ vm0, const unsigned* __restrict__ vm1,
        const unsigned* __restrict__ vm2, _Float16* __restrict__ finalh) {
    long tid = (long)blockIdx.x * 256 + threadIdx.x;
    int p = (int)(tid >> 6);
    int j = (int)(tid & 63);
    _Float16* frow = finalh + (long)p * KDIM;
    frow[ 64 + j] = (_Float16)__uint_as_float(vm0[(long)pv[p]            * 64 + j]);
    frow[192 + j] = (_Float16)__uint_as_float(vm1[(long)pv[NPTS + p]     * 64 + j]);
    frow[320 + j] = (_Float16)__uint_as_float(vm2[(long)pv[2 * NPTS + p] * 64 + j]);
}

// ---------------- AVFEO: WMMA GEMM [16,384]x[384,128] per block + epilogue ---
// Block = 256 threads = 8 waves; wave w computes cols [16w,16w+16); 12 K-steps.
__global__ void __launch_bounds__(256) hv_avfeo_wmma(
        const _Float16* __restrict__ finalh, const _Float16* __restrict__ packB,
        const float* __restrict__ pts, const int* __restrict__ pv3,
        const float* __restrict__ s3, const float* __restrict__ c3,
        const float* __restrict__ Watt2, unsigned* __restrict__ vm3) {
    __shared__ float attL[16][8];
    int rbase = blockIdx.x * 16;                 // 18750 blocks * 16 = 300000 exactly
    if (threadIdx.x < 16) {
        int p = rbase + threadIdx.x;
        float px = pts[4 * p + 0], py = pts[4 * p + 1];
        float pz = pts[4 * p + 2], pw = pts[4 * p + 3];
        int v = pv3[p];
        float inv = 1.0f / fmaxf(c3[v], 1.0f);
        float mx = s3[v * 4 + 0] * inv, my = s3[v * 4 + 1] * inv;
        float mz = s3[v * 4 + 2] * inv, mw = s3[v * 4 + 3] * inv;
        attL[threadIdx.x][0] = px - mx; attL[threadIdx.x][1] = py - my;
        attL[threadIdx.x][2] = pz - mz; attL[threadIdx.x][3] = pw;
        attL[threadIdx.x][4] = mx;      attL[threadIdx.x][5] = my;
        attL[threadIdx.x][6] = mz;      attL[threadIdx.x][7] = mw;
    }
    __syncthreads();
    int wave = threadIdx.x >> 5;
    int lane = threadIdx.x & 31;
    int hi   = lane >> 4;
    int col  = wave * 16 + (lane & 15);
    float wc[8];
#pragma unroll
    for (int q = 0; q < 8; q++) wc[q] = Watt2[q * 128 + col];

    v8f acc = {};
    // A layout (16-bit 16x32): lanes 0-15 row=lane, K in {0..7,16..23};
    // lanes 16-31 same rows, K in {8..15,24..31}. Two 16B chunks per lane.
    const _Float16* arow = finalh + (long)(rbase + (lane & 15)) * KDIM + (hi ? 8 : 0);
    const _Float16* bptr = packB + ((long)(wave * 12) * 32 + lane) * 16;
#pragma unroll
    for (int t = 0; t < 12; t++) {
        v8h alo = *(const v8h*)(arow + t * 32);
        v8h ahi = *(const v8h*)(arow + t * 32 + 16);
        v16h a;
#pragma unroll
        for (int e = 0; e < 8; e++) { a[e] = alo[e]; a[e + 8] = ahi[e]; }
        v16h b = *(const v16h*)(bptr + (long)t * 32 * 16);
        acc = __builtin_amdgcn_wmma_f32_16x16x32_f16(
                false, a, false, b, (short)0, acc, false, false);
    }
    // Epilogue: pf = relu(acc), af = relu(att@W_att2), sf = pf*af, segment max.
#pragma unroll
    for (int e = 0; e < 8; e++) {
        int rl  = e + (hi ? 8 : 0);               // D layout: VGPR e -> row e / 8+e
        int row = rbase + rl;
        float pf = fmaxf(acc[e], 0.0f);
        float af = 0.0f;
#pragma unroll
        for (int q = 0; q < 8; q++) af += attL[rl][q] * wc[q];
        af = fmaxf(af, 0.0f);
        float sf = pf * af;                       // >= 0
        int v = pv3[row];
        atomicMax(&vm3[(long)v * 128 + col], __float_as_uint(sf));
    }
}

// ---------------- scatter vmax3 -> dense [B,128,H,W] --------------------------
__global__ void __launch_bounds__(256) hv_scatter(
        const unsigned* __restrict__ vm3, const int* __restrict__ vfs,
        float* __restrict__ out) {
    int t = blockIdx.x * blockDim.x + threadIdx.x;   // exactly V3S*128 threads
    int v = t >> 7;
    int c = t & 127;
    int b = vfs[3 * v + 0], y = vfs[3 * v + 1], x = vfs[3 * v + 2];
    out[(((long)b * 128 + c) * HH + y) * WW + x] = __uint_as_float(vm3[(long)v * 128 + c]);
}

extern "C" void kernel_launch(void* const* d_in, const int* in_sizes, int n_in,
                              void* d_out, int out_size, void* d_ws, size_t ws_size,
                              hipStream_t stream) {
    (void)in_sizes; (void)n_in; (void)ws_size;
    const float* pts   = (const float*)d_in[0];   // [300000,4]
    const float* Wpt1  = (const float*)d_in[1];   // [4,64]
    const float* Watt1 = (const float*)d_in[2];   // [8,64]
    const float* Wpt2  = (const float*)d_in[3];   // [384,128]
    const float* Watt2 = (const float*)d_in[4];   // [8,128]
    const int*   pv    = (const int*)d_in[5];     // [4,300000]
    const int*   vfs   = (const int*)d_in[6];     // [120000,3]
    float* out = (float*)d_out;

    char* ws = (char*)d_ws;
    float*     s0  = (float*)(ws + OFF_SUMS0);
    float*     s1  = (float*)(ws + OFF_SUMS1);
    float*     s2  = (float*)(ws + OFF_SUMS2);
    float*     s3  = (float*)(ws + OFF_SUMS3);
    float*     c0  = (float*)(ws + OFF_CNT0);
    float*     c1  = (float*)(ws + OFF_CNT1);
    float*     c2  = (float*)(ws + OFF_CNT2);
    float*     c3  = (float*)(ws + OFF_CNT3);
    unsigned*  vm0 = (unsigned*)(ws + OFF_VMAX0);
    unsigned*  vm1 = (unsigned*)(ws + OFF_VMAX1);
    unsigned*  vm2 = (unsigned*)(ws + OFF_VMAX2);
    unsigned*  vm3 = (unsigned*)(ws + OFF_VMAX3);
    _Float16*  pB  = (_Float16*)(ws + OFF_PACKB);
    _Float16*  fh  = (_Float16*)(ws + OFF_FINAL);

    // 1) zero scratch accumulators (sums/cnts/vmax) and dense output
    {
        long n16 = (long)(ZERO_BYTES / 16);
        int blocks = (int)((n16 + 255) / 256);
        hv_zero_u4<<<blocks, 256, 0, stream>>>((uint4*)ws, n16);
        long o16 = (long)out_size / 4;           // floats -> uint4
        int ob = (int)((o16 + 255) / 256);
        hv_zero_u4<<<ob, 256, 0, stream>>>((uint4*)out, o16);
    }
    // 2) pack B fragments for WMMA
    hv_pack_b<<<192, 256, 0, stream>>>(Wpt2, pB);
    // 3) segment sums + counts, all scales
    hv_seg_sums<<<(NPTS + 255) / 256, 256, 0, stream>>>(
        (const float4*)pts, pv, s0, c0, s1, c1, s2, c2, s3, c3);
    // 4) fused AVFE (3 input scales): NPTS*64 threads
    hv_avfe<<<(NPTS * 64) / 256, 256, 0, stream>>>(
        pts, pv, Wpt1, Watt1, s0, c0, s1, c1, s2, c2, vm0, vm1, vm2, fh);
    // 5) gather per-voxel maxima into final feature rows
    hv_gather<<<(NPTS * 64) / 256, 256, 0, stream>>>(pv, vm0, vm1, vm2, fh);
    // 6) AVFEO: WMMA GEMM + attention epilogue + segment max
    hv_avfeo_wmma<<<NPTS / 16, 256, 0, stream>>>(
        fh, pB, pts, pv + 3 * NPTS, s3, c3, Watt2, vm3);
    // 7) scatter to dense NCHW output
    hv_scatter<<<(V3S * 128) / 256, 256, 0, stream>>>(vm3, vfs, out);
}
